// MoELayer_7181185319327
// MI455X (gfx1250) — compile-verified, matrix operand order checked by reading, not run on hardware
//
#include <hip/hip_runtime.h>
#include <hip/hip_bf16.h>
#include <math.h>

typedef __attribute__((ext_vector_type(16))) _Float16 v16h;
typedef __attribute__((ext_vector_type(8)))  _Float16 v8h;
typedef __attribute__((ext_vector_type(8)))  float    v8f;

#define BM 128
#define BN 128
#define BK 32
#define PITCH 56   // halves -> 112B row pitch: 16B aligned, 28-dword bank stride (16 distinct banks)

// ---------------- gate pipeline ----------------

__global__ __launch_bounds__(256) void mean_kernel(const float* __restrict__ inp,
                                                   float* __restrict__ mean) {
  __shared__ float red[256];
  size_t base = (size_t)blockIdx.x * 1024;            // blockIdx.x = b*512 + c
  float4 v = *(const float4*)(inp + base + threadIdx.x * 4);
  red[threadIdx.x] = v.x + v.y + v.z + v.w;
  __syncthreads();
  for (int s = 128; s > 0; s >>= 1) {
    if (threadIdx.x < s) red[threadIdx.x] += red[threadIdx.x + s];
    __syncthreads();
  }
  if (threadIdx.x == 0) mean[blockIdx.x] = red[0] * (1.0f / 1024.0f);
}

__global__ void gate_kernel(const float* __restrict__ mean, const float* __restrict__ Wg,
                            const float* __restrict__ bg, const float* __restrict__ kArr,
                            int* __restrict__ pair_e, float* __restrict__ pair_scale) {
  __shared__ float lg[64];
  int t = threadIdx.x;
  if (t < 64) {                                        // logits[b][e]
    int b = t >> 3, e = t & 7;
    float s = bg[e];
    for (int c = 0; c < 512; ++c) s += mean[b * 512 + c] * Wg[c * 8 + e];
    lg[t] = s;
  }
  __syncthreads();
  if (t < 8) {                                         // per-sample fp32 softmax + top-2
    int b = t;
    float w[8], mx = -1e30f;
    for (int e = 0; e < 8; ++e) mx = fmaxf(mx, lg[b * 8 + e]);
    float sum = 0.0f;
    for (int e = 0; e < 8; ++e) { w[e] = __expf(lg[b * 8 + e] - mx); sum += w[e]; }
    float inv = 1.0f / sum;
    int i0 = 0;
    for (int e = 1; e < 8; ++e) if (w[e] > w[i0]) i0 = e;          // lowest index on ties
    int i1 = (i0 == 0) ? 1 : 0;
    for (int e = 0; e < 8; ++e) if (e != i0 && w[e] > w[i1]) i1 = e;
    float kk = kArr[b];
    pair_e[2 * b + 0] = i0; pair_scale[2 * b + 0] = kk * w[i0] * inv;
    pair_e[2 * b + 1] = i1; pair_scale[2 * b + 1] = kk * w[i1] * inv;
  }
}

// ---------------- f32 -> f16 transpose-cast: src[s][R][Cc] -> dst[s][Cc][R] ----------------

__global__ __launch_bounds__(256) void transpose_cast_kernel(const float* __restrict__ src,
                                                             _Float16* __restrict__ dst,
                                                             int R, int Cc) {
  __shared__ float tile[32][33];
  int s = blockIdx.z;
  const float* S = src + (size_t)s * R * Cc;
  _Float16*    D = dst + (size_t)s * R * Cc;
  int c0 = blockIdx.x * 32, r0 = blockIdx.y * 32;
  int tx = threadIdx.x, ty = threadIdx.y;              // (32, 8)
  #pragma unroll
  for (int j = 0; j < 32; j += 8)
    tile[ty + j][tx] = S[(size_t)(r0 + ty + j) * Cc + (c0 + tx)];
  __syncthreads();
  #pragma unroll
  for (int j = 0; j < 32; j += 8)
    D[(size_t)(c0 + ty + j) * R + (r0 + tx)] = (_Float16)tile[tx][ty + j];
}

// ---------------- WMMA tile helpers ----------------

__device__ __forceinline__ v16h load_frag(const _Float16* sm, int row, int hi) {
  // ISA 16-bit fragment: lane = (row = m16 + lane&15, K-half = lane>>4);
  // elements 0..7 = K(hi*8 .. hi*8+7), elements 8..15 = K(16+hi*8 ..)
  const _Float16* p = sm + row * PITCH + hi * 8;
  v8h lo = *(const v8h*)(p);
  v8h hh = *(const v8h*)(p + 16);
  return __builtin_shufflevector(lo, hh, 0,1,2,3,4,5,6,7,8,9,10,11,12,13,14,15);
}

// global tile (128 rows x 32 halves) -> registers (two 16B chunks per thread)
__device__ __forceinline__ void load_tile_g2r(v8h r[2], const _Float16* g, int ldg, int tid) {
  #pragma unroll
  for (int i = 0; i < 2; ++i) {
    int chunk = tid + i * 256;
    int row = chunk >> 2;
    int col = (chunk & 3) * 8;
    r[i] = *(const v8h*)(g + (size_t)row * ldg + col);
  }
}

// registers -> LDS (padded pitch, 16B-aligned ds_store_b128)
__device__ __forceinline__ void store_tile_r2s(_Float16* sm, const v8h r[2], int tid) {
  #pragma unroll
  for (int i = 0; i < 2; ++i) {
    int chunk = tid + i * 256;
    int row = chunk >> 2;
    int col = (chunk & 3) * 8;
    *(v8h*)(sm + row * PITCH + col) = r[i];
  }
}

// ---------------- GEMM1: H = gelu(X^T @ W1 + b1), per (b,e) pair, f16 out ----------------

__global__ __launch_bounds__(256) void gemm1_kernel(const _Float16* __restrict__ Xh,
                                                    const _Float16* __restrict__ W1T,
                                                    const float* __restrict__ b1,
                                                    const int* __restrict__ pair_e,
                                                    _Float16* __restrict__ Hh) {
  __shared__ _Float16 As[BM * PITCH];
  __shared__ _Float16 Bs[BN * PITCH];
  const int pair = blockIdx.z;
  const int b = pair >> 1;
  const int e = pair_e[pair];
  const int Mblk = blockIdx.x * BM;                    // pixels
  const int Nblk = blockIdx.y * BN;                    // Dh
  const int tid = threadIdx.x, lane = tid & 31, wave = tid >> 5;
  const int hi = lane >> 4, ln = lane & 15;
  const int wm = (wave >> 1) * 32, wn = (wave & 1) * 64;

  const _Float16* Ag = Xh  + ((size_t)b << 19) + (size_t)Mblk * 512;
  const _Float16* Bg = W1T + ((size_t)e << 19) + (size_t)Nblk * 512;

  v8f acc[2][4] = {};
  v8h ra[2], rb[2];
  load_tile_g2r(ra, Ag, 512, tid);                     // prologue prefetch
  load_tile_g2r(rb, Bg, 512, tid);
  for (int k0 = 0; k0 < 512; k0 += BK) {
    __syncthreads();                                   // prev-iter frag reads done
    store_tile_r2s(As, ra, tid);
    store_tile_r2s(Bs, rb, tid);
    __syncthreads();                                   // tiles visible
    if (k0 + BK < 512) {                               // prefetch next K tile; overlaps WMMAs
      load_tile_g2r(ra, Ag + k0 + BK, 512, tid);
      load_tile_g2r(rb, Bg + k0 + BK, 512, tid);
    }
    v16h af[2], bf[4];
    #pragma unroll
    for (int im = 0; im < 2; ++im) af[im] = load_frag(As, wm + im * 16 + ln, hi);
    #pragma unroll
    for (int in = 0; in < 4; ++in) bf[in] = load_frag(Bs, wn + in * 16 + ln, hi);
    #pragma unroll
    for (int im = 0; im < 2; ++im)
      #pragma unroll
      for (int in = 0; in < 4; ++in)
        acc[im][in] = __builtin_amdgcn_wmma_f32_16x16x32_f16(
            false, af[im], false, bf[in], (short)0, acc[im][in], false, false);
  }
  // epilogue: +b1, exact gelu, store f16 H[pair][m][n]
  #pragma unroll
  for (int im = 0; im < 2; ++im) {
    const int subM = Mblk + wm + im * 16 + 8 * hi;
    #pragma unroll
    for (int in = 0; in < 4; ++in) {
      const int n = Nblk + wn + in * 16 + ln;
      const float b1v = b1[e * 1024 + n];
      _Float16* dst = Hh + ((size_t)pair << 20) + (size_t)subM * 1024 + n;
      #pragma unroll
      for (int r = 0; r < 8; ++r) {
        float x = acc[im][in][r] + b1v;
        float g = 0.5f * x * (1.0f + erff(x * 0.70710678118654752f));
        dst[(size_t)r * 1024] = (_Float16)g;
      }
    }
  }
}

// ------- GEMM2: out = inputs + sum_j s_j * (H_j @ W2_j + b2_j), fused residual -------

__global__ __launch_bounds__(256) void gemm2_kernel(const _Float16* __restrict__ Hh,
                                                    const _Float16* __restrict__ W2T,
                                                    const float* __restrict__ b2,
                                                    const int* __restrict__ pair_e,
                                                    const float* __restrict__ pair_scale,
                                                    const float* __restrict__ inp,
                                                    float* __restrict__ out) {
  __shared__ _Float16 As[BM * PITCH];
  __shared__ _Float16 Bs[BN * PITCH];
  const int b = blockIdx.z;
  const int Mblk = blockIdx.x * BM;                    // pixels
  const int Nblk = blockIdx.y * BN;                    // channels
  const int tid = threadIdx.x, lane = tid & 31, wave = tid >> 5;
  const int hi = lane >> 4, ln = lane & 15;
  const int wm = (wave >> 1) * 32, wn = (wave & 1) * 64;

  v8f accT[2][4] = {};
  for (int j = 0; j < 2; ++j) {                        // this sample's two experts
    const int pair = b * 2 + j;
    const int e = pair_e[pair];
    const float s = pair_scale[pair];
    const _Float16* Ag = Hh  + ((size_t)pair << 20) + (size_t)Mblk * 1024;
    const _Float16* Bg = W2T + ((size_t)e << 19) + (size_t)Nblk * 1024;
    v8f acc[2][4] = {};
    v8h ra[2], rb[2];
    load_tile_g2r(ra, Ag, 1024, tid);                  // per-expert prologue prefetch
    load_tile_g2r(rb, Bg, 1024, tid);
    for (int k0 = 0; k0 < 1024; k0 += BK) {
      __syncthreads();
      store_tile_r2s(As, ra, tid);
      store_tile_r2s(Bs, rb, tid);
      __syncthreads();
      if (k0 + BK < 1024) {                            // prefetch next K tile; overlaps WMMAs
        load_tile_g2r(ra, Ag + k0 + BK, 1024, tid);
        load_tile_g2r(rb, Bg + k0 + BK, 1024, tid);
      }
      v16h af[2], bf[4];
      #pragma unroll
      for (int im = 0; im < 2; ++im) af[im] = load_frag(As, wm + im * 16 + ln, hi);
      #pragma unroll
      for (int in = 0; in < 4; ++in) bf[in] = load_frag(Bs, wn + in * 16 + ln, hi);
      #pragma unroll
      for (int im = 0; im < 2; ++im)
        #pragma unroll
        for (int in = 0; in < 4; ++in)
          acc[im][in] = __builtin_amdgcn_wmma_f32_16x16x32_f16(
              false, af[im], false, bf[in], (short)0, acc[im][in], false, false);
    }
    #pragma unroll
    for (int im = 0; im < 2; ++im)
      #pragma unroll
      for (int in = 0; in < 4; ++in) {
        const int n = Nblk + wn + in * 16 + ln;
        const float b2v = b2[e * 512 + n];
        accT[im][in] += s * (acc[im][in] + b2v);
      }
  }
  // residual add + store: out[b][c=n][p=m]; each lane owns 8 consecutive pixels
  #pragma unroll
  for (int im = 0; im < 2; ++im) {
    const int m0 = Mblk + wm + im * 16 + 8 * hi;
    #pragma unroll
    for (int in = 0; in < 4; ++in) {
      const int n = Nblk + wn + in * 16 + ln;
      size_t base = ((size_t)b << 19) + (size_t)n * 1024 + m0;
      float4 r0 = *(const float4*)(inp + base);
      float4 r1 = *(const float4*)(inp + base + 4);
      v8f t2 = accT[im][in];
      r0.x += t2[0]; r0.y += t2[1]; r0.z += t2[2]; r0.w += t2[3];
      r1.x += t2[4]; r1.y += t2[5]; r1.z += t2[6]; r1.w += t2[7];
      *(float4*)(out + base)     = r0;
      *(float4*)(out + base + 4) = r1;
    }
  }
}

// ---------------- host ----------------

extern "C" void kernel_launch(void* const* d_in, const int* in_sizes, int n_in,
                              void* d_out, int out_size, void* d_ws, size_t ws_size,
                              hipStream_t stream) {
  (void)in_sizes; (void)n_in; (void)out_size; (void)ws_size;
  const float* inputs = (const float*)d_in[0];
  const float* kArr   = (const float*)d_in[1];
  const float* Wg     = (const float*)d_in[2];
  const float* bg     = (const float*)d_in[3];
  const float* W1     = (const float*)d_in[4];
  const float* b1     = (const float*)d_in[5];
  const float* W2     = (const float*)d_in[6];
  const float* b2     = (const float*)d_in[7];
  float* out = (float*)d_out;

  char* ws = (char*)d_ws;
  float*    mean       = (float*)(ws);                            // 16 KB
  int*      pair_e     = (int*)(ws + 16384);                      // 64 B
  float*    pair_scale = (float*)(ws + 16448);                    // 64 B
  _Float16* Xh   = (_Float16*)(ws + 32768);                       // 8 MB [b][p][c]
  _Float16* W1T  = (_Float16*)(ws + 32768 + 1u * 8388608);        // 8 MB [e][d][c]
  _Float16* W2T  = (_Float16*)(ws + 32768 + 2u * 8388608);        // 8 MB [e][c][d]
  _Float16* Hh   = (_Float16*)(ws + 32768 + 3u * 8388608);        // 32 MB [pair][p][d]

  hipLaunchKernelGGL(mean_kernel, dim3(4096), dim3(256), 0, stream, inputs, mean);
  hipLaunchKernelGGL(gate_kernel, dim3(1), dim3(64), 0, stream,
                     mean, Wg, bg, kArr, pair_e, pair_scale);
  hipLaunchKernelGGL(transpose_cast_kernel, dim3(32, 16, 8), dim3(32, 8), 0, stream,
                     inputs, Xh, 512, 1024);
  hipLaunchKernelGGL(transpose_cast_kernel, dim3(32, 16, 8), dim3(32, 8), 0, stream,
                     W1, W1T, 512, 1024);
  hipLaunchKernelGGL(transpose_cast_kernel, dim3(16, 32, 8), dim3(32, 8), 0, stream,
                     W2, W2T, 1024, 512);
  hipLaunchKernelGGL(gemm1_kernel, dim3(8, 8, 16), dim3(256), 0, stream,
                     Xh, W1T, b1, pair_e, Hh);
  hipLaunchKernelGGL(gemm2_kernel, dim3(8, 4, 8), dim3(256), 0, stream,
                     Hh, W2T, b2, pair_e, pair_scale, inputs, out);
}
